// GCN_graph_5866925326853
// MI455X (gfx1250) — compile-verified, matrix operand order checked by reading, not run on hardware
//
#include <hip/hip_runtime.h>
#include <hip/hip_bf16.h>

typedef __attribute__((ext_vector_type(16))) __bf16 v16bf;
typedef __attribute__((ext_vector_type(8)))  __bf16 v8bf;
typedef __attribute__((ext_vector_type(8)))  float  v8f;

#define HDIM 512

// ---------------- f32 -> bf16 convert ----------------
__global__ __launch_bounds__(256) void cvt_f32_bf16(const float* __restrict__ in,
                                                    __bf16* __restrict__ out, int n) {
    int i = blockIdx.x * 256 + threadIdx.x;
    if (i < n) out[i] = (__bf16)in[i];
}

// ---------------- W[k][n] (f32) -> Wt[n][k] (bf16) ----------------
__global__ __launch_bounds__(256) void transpose_cvt(const float* __restrict__ W,
                                                     __bf16* __restrict__ Wt) {
    int k = blockIdx.x * 256 + threadIdx.x;   // 0..511
    int n = blockIdx.y;                       // 0..511
    Wt[n * HDIM + k] = (__bf16)W[k * HDIM + n];
}

// ---------------- GEMM: S[M x 512] = A[M x 512]bf16 * Wt^T, f32 accumulate ----------------
// Wave computes a 16(M) x 64(N) strip via 4 accumulators of v_wmma_f32_16x16x32_bf16.
__global__ __launch_bounds__(256) void gemm_bf16_wmma(const __bf16* __restrict__ A,
                                                      const __bf16* __restrict__ Wt,
                                                      float* __restrict__ S, int M) {
    const int lane = threadIdx.x & 31;
    const int wave = threadIdx.x >> 5;
    const int half = lane >> 4;      // 0: lanes 0-15, 1: lanes 16-31
    const int l15  = lane & 15;
    const int m0   = (blockIdx.x * 8 + wave) * 16;
    if (m0 >= M) return;             // wave-uniform: EXEC stays all-1s for WMMA
    const int n0   = blockIdx.y * 64;

    v8f acc[4] = {};
    const __bf16* __restrict__ arow = A + (size_t)(m0 + l15) * HDIM;

    for (int k0 = 0; k0 < HDIM; k0 += 32) {
        const int ka = k0 + half * 8;
        // A fragment: elems 0..7 = K[ka..ka+7], elems 8..15 = K[ka+16..ka+23]
        v8bf a_lo = *(const v8bf*)(arow + ka);
        v8bf a_hi = *(const v8bf*)(arow + ka + 16);
        __builtin_prefetch(arow + ka + 32, 0, 1);  // next K-slab
        v16bf afrag;
        #pragma unroll
        for (int i = 0; i < 8; ++i) { afrag[i] = a_lo[i]; afrag[i + 8] = a_hi[i]; }

        #pragma unroll
        for (int t = 0; t < 4; ++t) {
            const __bf16* __restrict__ brow = Wt + (size_t)(n0 + t * 16 + l15) * HDIM;
            v8bf b_lo = *(const v8bf*)(brow + ka);
            v8bf b_hi = *(const v8bf*)(brow + ka + 16);
            v16bf bfrag;
            #pragma unroll
            for (int i = 0; i < 8; ++i) { bfrag[i] = b_lo[i]; bfrag[i + 8] = b_hi[i]; }
            acc[t] = __builtin_amdgcn_wmma_f32_16x16x32_bf16(
                false, afrag, false, bfrag, (short)0, acc[t], false, false);
        }
    }

    // C/D layout: lane column = n0 + t*16 + l15 ; VGPR r -> row m0 + half*8 + r
    #pragma unroll
    for (int t = 0; t < 4; ++t) {
        const int ncol = n0 + t * 16 + l15;
        #pragma unroll
        for (int r = 0; r < 8; ++r) {
            S[(size_t)(m0 + half * 8 + r) * HDIM + ncol] = acc[t][r];
        }
    }
}

// ---------------- COO scatter-add: agg[rows[e]] += vals[e] * S[cols[e]] ----------------
// 128 threads per edge, float4 per thread, global f32 atomics.
__global__ __launch_bounds__(256) void scatter_edges(const int* __restrict__ rows,
                                                     const int* __restrict__ cols,
                                                     const float* __restrict__ vals,
                                                     const float* __restrict__ S,
                                                     float* __restrict__ agg, int E) {
    long long t = (long long)blockIdx.x * 256 + threadIdx.x;
    int e = (int)(t >> 7);
    if (e >= E) return;
    int j = ((int)t & 127) << 2;
    float a = vals[e];
    const float4 sv = *(const float4*)(S + (size_t)cols[e] * HDIM + j);
    float* dst = agg + (size_t)rows[e] * HDIM + j;
    atomicAdd(dst + 0, a * sv.x);
    atomicAdd(dst + 1, a * sv.y);
    atomicAdd(dst + 2, a * sv.z);
    atomicAdd(dst + 3, a * sv.w);
}

// ---------------- fused bias + ReLU + bf16 convert + max/sum pooling ----------------
// Block covers a 128-row strip; thread owns columns (tid) and (tid+256).
__global__ __launch_bounds__(256) void bias_relu_pool(const float* __restrict__ agg,
                                                      const float* __restrict__ b,
                                                      __bf16* __restrict__ hout,
                                                      unsigned* __restrict__ maxbuf,
                                                      float* __restrict__ sumbuf, int M) {
    const int c0 = threadIdx.x;
    const int c1 = c0 + 256;
    const int r0 = blockIdx.x * 128;
    const int rend = min(r0 + 128, M);
    const float b0 = b[c0], b1 = b[c1];
    float mx0 = 0.f, mx1 = 0.f, sm0 = 0.f, sm1 = 0.f;
    for (int r = r0; r < rend; ++r) {
        const size_t base = (size_t)r * HDIM;
        float v0 = agg[base + c0] + b0; v0 = v0 > 0.f ? v0 : 0.f;
        float v1 = agg[base + c1] + b1; v1 = v1 > 0.f ? v1 : 0.f;
        hout[base + c0] = (__bf16)v0;
        hout[base + c1] = (__bf16)v1;
        mx0 = fmaxf(mx0, v0); mx1 = fmaxf(mx1, v1);
        sm0 += v0; sm1 += v1;
    }
    // ReLU output >= 0, so uint bit order == float order; init bits 0 == 0.0f
    atomicMax(&maxbuf[c0], __float_as_uint(mx0));
    atomicMax(&maxbuf[c1], __float_as_uint(mx1));
    atomicAdd(&sumbuf[c0], sm0);
    atomicAdd(&sumbuf[c1], sm1);
}

// ---------------- head MLP + log_softmax, single block ----------------
__global__ __launch_bounds__(256) void head_mlp(const unsigned* __restrict__ maxbuf,
                                                const float* __restrict__ sumbuf,
                                                const float* __restrict__ l1W,
                                                const float* __restrict__ l1b,
                                                const float* __restrict__ l2W,
                                                const float* __restrict__ l2b,
                                                const float* __restrict__ l3W,
                                                const float* __restrict__ l3b,
                                                float* __restrict__ out, float invN) {
    __shared__ float g[1024];
    __shared__ float z1[128];
    __shared__ float z2[64];
    __shared__ float z3[10];
    const int tid = threadIdx.x;

    // g = sum over 3 layers of [max | mean]
    for (int j = tid; j < 512; j += 256) {
        g[j] = __uint_as_float(maxbuf[j]) + __uint_as_float(maxbuf[512 + j]) +
               __uint_as_float(maxbuf[1024 + j]);
        g[512 + j] = (sumbuf[j] + sumbuf[512 + j] + sumbuf[1024 + j]) * invN;
    }
    __syncthreads();

    if (tid < 128) {
        float acc = l1b[tid];
        for (int i = 0; i < 1024; ++i) acc += g[i] * l1W[i * 128 + tid];
        z1[tid] = acc > 0.f ? acc : 0.f;
    }
    __syncthreads();

    if (tid < 64) {
        float acc = l2b[tid];
        for (int i = 0; i < 128; ++i) acc += z1[i] * l2W[i * 64 + tid];
        z2[tid] = acc > 0.f ? acc : 0.f;
    }
    __syncthreads();

    if (tid < 10) {
        float acc = l3b[tid];
        for (int i = 0; i < 64; ++i) acc += z2[i] * l3W[i * 10 + tid];
        z3[tid] = acc;
    }
    __syncthreads();

    if (tid == 0) {
        float m = z3[0];
        for (int c = 1; c < 10; ++c) m = fmaxf(m, z3[c]);
        float s = 0.f;
        for (int c = 0; c < 10; ++c) s += __expf(z3[c] - m);
        float lse = m + logf(s);
        for (int c = 0; c < 10; ++c) out[c] = z3[c] - lse;
    }
}

extern "C" void kernel_launch(void* const* d_in, const int* in_sizes, int n_in,
                              void* d_out, int out_size, void* d_ws, size_t ws_size,
                              hipStream_t stream) {
    const float* x    = (const float*)d_in[0];
    const int*   rows = (const int*)d_in[1];
    const int*   cols = (const int*)d_in[2];
    const float* av   = (const float*)d_in[3];
    const float* W[3] = {(const float*)d_in[4], (const float*)d_in[6], (const float*)d_in[8]};
    const float* b[3] = {(const float*)d_in[5], (const float*)d_in[7], (const float*)d_in[9]};
    const float* l1W = (const float*)d_in[10];
    const float* l1b = (const float*)d_in[11];
    const float* l2W = (const float*)d_in[12];
    const float* l2b = (const float*)d_in[13];
    const float* l3W = (const float*)d_in[14];
    const float* l3b = (const float*)d_in[15];
    float* out = (float*)d_out;

    const int M = in_sizes[0] / HDIM;   // 20000 nodes
    const int E = in_sizes[1];          // 320000 edges
    const size_t elems = (size_t)M * HDIM;

    // workspace layout (256B aligned)
    char* ws = (char*)d_ws;
    __bf16*   h_bf16 = (__bf16*)ws;                          // M*512*2
    float*    S      = (float*)(ws + 20480000);              // M*512*4
    float*    agg    = (float*)(ws + 20480000 + 40960000);   // M*512*4
    __bf16*   Wt     = (__bf16*)(ws + 102400000);            // 3 * 512*512*2
    unsigned* maxbuf = (unsigned*)(ws + 103972864);          // 3*512 u32
    float*    sumbuf = (float*)(ws + 103979008);             // 3*512 f32

    // zero pooling buffers (0 bits == 0.0f for both)
    hipMemsetAsync(maxbuf, 0, 2 * 3 * HDIM * sizeof(float), stream);

    // weights: transpose+convert to bf16 once per call
    for (int L = 0; L < 3; ++L)
        transpose_cvt<<<dim3(HDIM / 256, HDIM), 256, 0, stream>>>(W[L], Wt + (size_t)L * HDIM * HDIM);

    // x -> bf16
    cvt_f32_bf16<<<(int)((elems + 255) / 256), 256, 0, stream>>>(x, h_bf16, (int)elems);

    const dim3 gemmGrid((M + 127) / 128, HDIM / 64);
    const int scatterBlocks = (int)(((long long)E * 128 + 255) / 256);

    for (int L = 0; L < 3; ++L) {
        gemm_bf16_wmma<<<gemmGrid, 256, 0, stream>>>(h_bf16, Wt + (size_t)L * HDIM * HDIM, S, M);
        hipMemsetAsync(agg, 0, elems * sizeof(float), stream);
        scatter_edges<<<scatterBlocks, 256, 0, stream>>>(rows, cols, av, S, agg, E);
        bias_relu_pool<<<(M + 127) / 128, 256, 0, stream>>>(agg, b[L], h_bf16,
                                                            maxbuf + L * HDIM, sumbuf + L * HDIM, M);
    }

    head_mlp<<<1, 256, 0, stream>>>(maxbuf, sumbuf, l1W, l1b, l2W, l2b, l3W, l3b,
                                    out, 1.0f / (float)M);
}
